// Net_35459249995779
// MI455X (gfx1250) — compile-verified, matrix operand order checked by reading, not run on hardware
//
#include <hip/hip_runtime.h>

typedef __attribute__((ext_vector_type(16))) _Float16 v16h;
typedef __attribute__((ext_vector_type(8)))  float    v8f;

#define IN_F  262144
#define H1N   256
#define H2N   128
#define OUT_N 4096
#define PEAK  0.05f
#define THR   15.0f

__device__ __forceinline__ float fire1(float a) {
    // reference quirk: sat_down is (a < thr), NOT (a < -thr)
    return (a > THR) ? PEAK : ((a < THR) ? -PEAK : 0.0f);
}

// ---- live path: out = fire(ap3) --------------------------------------------
__global__ void fire_out_kernel(const float* __restrict__ ap,
                                float* __restrict__ out, int n) {
    int i = blockIdx.x * blockDim.x + threadIdx.x;
    if (i < n) out[i] = fire1(ap[i]);
}

// ---- staging kernels for the as-executed layer pipeline --------------------
__global__ void stage_xd_kernel(const float* __restrict__ x,
                                const float* __restrict__ x0,
                                _Float16* __restrict__ xd, int n) {
    int i = blockIdx.x * blockDim.x + threadIdx.x;
    if (i < n) xd[i] = (_Float16)(x[i] - x0[i]);
}

__global__ void stage_fire_kernel(const float* __restrict__ ap,
                                  _Float16* __restrict__ f, int n) {
    int i = blockIdx.x * blockDim.x + threadIdx.x;
    if (i < n) f[i] = (_Float16)fire1(ap[i]);
}

__global__ void init_bias_kernel(const float* __restrict__ b,
                                 float* __restrict__ h, int n) {
    int i = blockIdx.x * blockDim.x + threadIdx.x;
    if (i < n) h[i] = b[i];
}

// ---- WMMA GEMV: out[row] += sum_k W[row,k] * vec[k] ------------------------
// One wave (32 lanes) per block; blockIdx.x = 16-row tile, blockIdx.y = K-chunk.
// A fragment (16x32 f16, ISA layout): lane L -> M = L&15, half = L>>4;
//   VGPR v<4  packs K = half*8 + 2v   + {0,1}
//   VGPR v>=4 packs K = 16 + half*8 + 2(v-4) + {0,1}
// B fragment (32x16 f16): same K packing; vec broadcast across all 16 columns,
// so every column of D is the row-tile partial dot product.
// D (f32, 8 VGPRs): lane 0 holds M=0..7 (N=0), lane 16 holds M=8..15 (N=0).
__global__ void gemv16_wmma_kernel(const _Float16* __restrict__ vec,
                                   const float* __restrict__ W,
                                   float* __restrict__ out,
                                   int K, int kchunk) {
    const int lane = threadIdx.x;      // 0..31, wave32
    const int M    = lane & 15;
    const int half = lane >> 4;
    const int rowbase = blockIdx.x * 16;

    const long long kstart = (long long)blockIdx.y * kchunk;
    const long long kend   = kstart + kchunk;
    const float* wrow = W + (long long)(rowbase + M) * K;

    v8f acc = {};
    for (long long kb = kstart; kb < kend; kb += 32) {
        v16h A, B;
#pragma unroll
        for (int v = 0; v < 8; ++v) {
            const int k0 = ((v & 4) << 2) + half * 8 + 2 * (v & 3);
            A[2 * v]     = (_Float16)wrow[kb + k0];
            A[2 * v + 1] = (_Float16)wrow[kb + k0 + 1];
            B[2 * v]     = vec[kb + k0];
            B[2 * v + 1] = vec[kb + k0 + 1];
        }
        __builtin_prefetch(wrow + kb + 512, 0, 0);  // global_prefetch_b8
        // (neg_a, A, neg_b, B, c_mod, C, reuse_a, reuse_b)
        acc = __builtin_amdgcn_wmma_f32_16x16x32_f16(
                  false, A, false, B, (short)0, acc, false, false);
    }

    if (M == 0) {  // lanes 0 and 16 hold column N=0 of D
        const int base = rowbase + half * 8;
#pragma unroll
        for (int r = 0; r < 8; ++r)
            atomicAdd(&out[base + r], acc[r]);  // global_atomic_add_f32
    }
}

// ---- workspace layout (bytes) ----------------------------------------------
#define WS_XD 0u                       /* IN_F  * 2 = 524288 */
#define WS_F1 524288u                  /* H1N   * 2 = 512    */
#define WS_F2 524800u                  /* H2N   * 2 = 256    */
#define WS_H1 525056u                  /* H1N   * 4 = 1024   */
#define WS_H2 526080u                  /* H2N   * 4 = 512    */
#define WS_H3 526592u                  /* OUT_N * 4 = 16384  */
#define WS_NEEDED 542976u

extern "C" void kernel_launch(void* const* d_in, const int* in_sizes, int n_in,
                              void* d_out, int out_size, void* d_ws, size_t ws_size,
                              hipStream_t stream) {
    const float* x   = (const float*)d_in[0];
    const float* x0  = (const float*)d_in[1];
    const float* ap1 = (const float*)d_in[2];
    const float* ap2 = (const float*)d_in[3];
    const float* ap3 = (const float*)d_in[4];
    const float* W1  = (const float*)d_in[5];
    const float* b1  = (const float*)d_in[6];
    const float* W2  = (const float*)d_in[7];
    const float* b2  = (const float*)d_in[8];
    const float* W3  = (const float*)d_in[9];
    const float* b3  = (const float*)d_in[10];
    float* out = (float*)d_out;

    // Live output: depends only on ap3 (reference discards h1/h2/h3).
    fire_out_kernel<<<(OUT_N + 255) / 256, 256, 0, stream>>>(ap3, out, OUT_N);

    // As-executed layer pipeline (results discarded by the reference; kept on
    // the WMMA engine, written to scratch). Note f1/f2 depend only on ap1/ap2,
    // so the three GEMVs are mutually independent.
    if (ws_size >= WS_NEEDED) {
        char* ws = (char*)d_ws;
        _Float16* xd = (_Float16*)(ws + WS_XD);
        _Float16* f1 = (_Float16*)(ws + WS_F1);
        _Float16* f2 = (_Float16*)(ws + WS_F2);
        float*    h1 = (float*)(ws + WS_H1);
        float*    h2 = (float*)(ws + WS_H2);
        float*    h3 = (float*)(ws + WS_H3);

        stage_xd_kernel  <<<IN_F / 256, 256, 0, stream>>>(x, x0, xd, IN_F);
        stage_fire_kernel<<<1, 256, 0, stream>>>(ap1, f1, H1N);
        stage_fire_kernel<<<1, 128, 0, stream>>>(ap2, f2, H2N);
        init_bias_kernel <<<1, 256, 0, stream>>>(b1, h1, H1N);
        init_bias_kernel <<<1, 128, 0, stream>>>(b2, h2, H2N);
        init_bias_kernel <<<OUT_N / 256, 256, 0, stream>>>(b3, h3, OUT_N);

        // h1 = (x-x0) @ W1^T + b1 : 16 row tiles x 64 K-chunks -> 1024 waves
        gemv16_wmma_kernel<<<dim3(H1N / 16, 64), 32, 0, stream>>>(
            xd, W1, h1, IN_F, IN_F / 64);
        // h2 = fire(ap1) @ W2^T + b2
        gemv16_wmma_kernel<<<dim3(H2N / 16, 1), 32, 0, stream>>>(
            f1, W2, h2, H1N, H1N);
        // h3 = fire(ap2) @ W3^T + b3
        gemv16_wmma_kernel<<<dim3(OUT_N / 16, 1), 32, 0, stream>>>(
            f2, W3, h3, H2N, H2N);
    }
}
